// DbrxMoe_78331613545212
// MI455X (gfx1250) — matrix operand order repaired and checked by reading, not compiled, not run on hardware
//
#include <hip/hip_runtime.h>

#define T_TOK 2048
#define HDIM  1024
#define FDIM  1024
#define NEXP  8

#define BM 128   // block tile rows (tokens)
#define BN 64    // block tile cols
#define KS 32    // K-step (one bf16 WMMA K)

typedef __bf16 bf16_t;
typedef __attribute__((ext_vector_type(16))) __bf16 v16bf;
typedef __attribute__((ext_vector_type(8)))  __bf16 v8bf;
typedef __attribute__((ext_vector_type(8)))  float  v8f;
typedef __attribute__((ext_vector_type(4)))  float  v4f;
typedef __attribute__((ext_vector_type(4)))  unsigned v4u;

// ---------------------------------------------------------------------------
// packed f32x2 -> bf16x2 (RNE) as one u32: integer ops only, no b16 moves
// ---------------------------------------------------------------------------
static __device__ __forceinline__ unsigned pack2bf(float lo, float hi) {
  unsigned ul = __builtin_bit_cast(unsigned, lo);
  unsigned uh = __builtin_bit_cast(unsigned, hi);
  ul = ul + 0x7FFFu + ((ul >> 16) & 1u);
  uh = uh + 0x7FFFu + ((uh >> 16) & 1u);
  return (uh & 0xFFFF0000u) | (ul >> 16);
}

static __device__ __forceinline__ bf16_t f2bf(float f) {
  unsigned u = __builtin_bit_cast(unsigned, f);
  unsigned short s = (unsigned short)((u + 0x7FFFu + ((u >> 16) & 1u)) >> 16);
  return __builtin_bit_cast(bf16_t, s);
}

// ---------------------------------------------------------------------------
// CDNA5 async global->LDS copy (GLOBAL_LOAD_ASYNC_TO_LDS_B128, tracked by
// ASYNCcnt) + wait. LDS operand is the low 32 bits of the flat pointer
// (= wave-relative LDS byte address for AS3-backed objects).
// ---------------------------------------------------------------------------
static __device__ __forceinline__ void async_copy_b128(const bf16_t* gptr, bf16_t* lptr) {
  unsigned l = (unsigned)(unsigned long long)lptr;
  asm volatile("global_load_async_to_lds_b128 %0, %1, off"
               :: "v"(l), "v"(gptr) : "memory");
}

static __device__ __forceinline__ void wait_async() {
  asm volatile("s_wait_asynccnt 0x0" ::: "memory");
}

// ---------------------------------------------------------------------------
// WMMA fragments out of LDS tiles (K-contiguous, row stride = KS halves).
// A (16x32): lanes 0-15 row M=lane, K-chunks {kb..kb+7, kb+16..kb+23}, kb=0;
//            lanes 16-31 same rows, kb=8.  -> two ds_load_b128 per lane.
// B (32x16): lane holds col N=lane&15; same K-chunks as A (kb=0 for lanes
//            0-15, kb=8 for lanes 16-31) -> two ds_load_b128.
// ---------------------------------------------------------------------------
static __device__ __forceinline__ v16bf lds_a_frag(const bf16_t* s, int m, int lane) {
  const int row = m + (lane & 15);
  const int kb  = (lane >> 4) << 3;
  const bf16_t* p = s + row * KS + kb;
  v8bf lo = *reinterpret_cast<const v8bf*>(p);
  v8bf hi = *reinterpret_cast<const v8bf*>(p + 16);
  return __builtin_shufflevector(lo, hi, 0,1,2,3,4,5,6,7,8,9,10,11,12,13,14,15);
}

static __device__ __forceinline__ v16bf lds_b_frag(const bf16_t* s, int n, int lane) {
  const int col = n + (lane & 15);
  const int kb  = (lane >> 4) << 3;
  const bf16_t* p = s + col * KS + kb;
  v8bf lo = *reinterpret_cast<const v8bf*>(p);
  v8bf hi = *reinterpret_cast<const v8bf*>(p + 16);
  return __builtin_shufflevector(lo, hi, 0,1,2,3,4,5,6,7,8,9,10,11,12,13,14,15);
}

static __device__ __forceinline__ v8f wmma_bf16(v16bf a, v16bf b, v8f c) {
  return __builtin_amdgcn_wmma_f32_16x16x32_bf16(false, a, false, b, (short)0, c,
                                                 false, false);
}

// ---------------------------------------------------------------------------
// Bulk f32 -> bf16 converter (8 elems/thread, packed u32 stores).
// Used for x once, and for each expert's weights (read-once streams).
// ---------------------------------------------------------------------------
__global__ __launch_bounds__(256) void f32_to_bf16_vec(const float* __restrict__ src,
                                                       bf16_t* __restrict__ dst) {
  const size_t i = ((size_t)blockIdx.x * blockDim.x + threadIdx.x) * 8;
  v4f a = *reinterpret_cast<const v4f*>(src + i);
  v4f b = *reinterpret_cast<const v4f*>(src + i + 4);
  v4u o;
  o[0] = pack2bf(a[0], a[1]);
  o[1] = pack2bf(a[2], a[3]);
  o[2] = pack2bf(b[0], b[1]);
  o[3] = pack2bf(b[2], b[3]);
  *reinterpret_cast<v4u*>(dst + i) = o;
}

__global__ __launch_bounds__(256) void zero_f32(float* __restrict__ p) {
  const int i = (blockIdx.x * blockDim.x + threadIdx.x) * 4;
  v4f z = {0.0f, 0.0f, 0.0f, 0.0f};
  *reinterpret_cast<v4f*>(p + i) = z;
}

// ---------------------------------------------------------------------------
// Router: logits -> softmax -> top-2 -> dense combine[T, E]
// ---------------------------------------------------------------------------
__global__ __launch_bounds__(256) void moe_router(const float* __restrict__ x,
                                                  const float* __restrict__ gate_w,
                                                  float* __restrict__ combine) {
  const int t = blockIdx.x * blockDim.x + threadIdx.x;
  if (t >= T_TOK) return;
  const float* __restrict__ xr = x + (size_t)t * HDIM;

  float logit[NEXP];
#pragma unroll
  for (int e = 0; e < NEXP; ++e) logit[e] = 0.0f;

  for (int h = 0; h < HDIM; h += 4) {
    v4f xv = *reinterpret_cast<const v4f*>(xr + h);
#pragma unroll
    for (int e = 0; e < NEXP; ++e) {
      v4f gv = *reinterpret_cast<const v4f*>(gate_w + (size_t)e * HDIM + h);
      logit[e] = fmaf(xv[0], gv[0],
                 fmaf(xv[1], gv[1],
                 fmaf(xv[2], gv[2],
                 fmaf(xv[3], gv[3], logit[e]))));
    }
  }

  float mx = logit[0];
#pragma unroll
  for (int e = 1; e < NEXP; ++e) mx = fmaxf(mx, logit[e]);
  float p[NEXP];
  float den = 0.0f;
#pragma unroll
  for (int e = 0; e < NEXP; ++e) { p[e] = __expf(logit[e] - mx); den += p[e]; }

  int i1 = 0;
#pragma unroll
  for (int e = 1; e < NEXP; ++e) if (p[e] > p[i1]) i1 = e;
  int i2 = (i1 == 0) ? 1 : 0;
#pragma unroll
  for (int e = 0; e < NEXP; ++e) if (e != i1 && p[e] > p[i2]) i2 = e;

  const float inv = __builtin_amdgcn_rcpf(den);
#pragma unroll
  for (int e = 0; e < NEXP; ++e)
    combine[(size_t)t * NEXP + e] = (e == i1 || e == i2) ? p[e] * inv : 0.0f;
}

// ---------------------------------------------------------------------------
// Async tile staging. Thread tid covers 16B chunks: tiles are rows x 64B
// (KS=32 bf16), i.e. 4 chunks per row.
// ---------------------------------------------------------------------------
static __device__ __forceinline__ void stage_ffn1(const bf16_t* __restrict__ xb,
                                                  const bf16_t* __restrict__ wg,
                                                  const bf16_t* __restrict__ wv,
                                                  bf16_t* sA, bf16_t* sBg, bf16_t* sBv,
                                                  int m_blk, int n_blk, int k0, int tid) {
  const int r = tid >> 2;             // 0..63
  const int c = (tid & 3) * 8;        // half-element offset within K-slice
  async_copy_b128(xb + (size_t)(m_blk + r) * HDIM + k0 + c,      sA  + r * KS + c);
  async_copy_b128(xb + (size_t)(m_blk + 64 + r) * HDIM + k0 + c, sA  + (64 + r) * KS + c);
  async_copy_b128(wg + (size_t)(n_blk + r) * HDIM + k0 + c,      sBg + r * KS + c);
  async_copy_b128(wv + (size_t)(n_blk + r) * HDIM + k0 + c,      sBv + r * KS + c);
}

static __device__ __forceinline__ void stage_ffn2(const bf16_t* __restrict__ hid,
                                                  const bf16_t* __restrict__ w2b,
                                                  bf16_t* sA, bf16_t* sB,
                                                  int m_blk, int n_blk, int k0, int tid) {
  const int r = tid >> 2;
  const int c = (tid & 3) * 8;
  async_copy_b128(hid + (size_t)(m_blk + r) * FDIM + k0 + c,      sA + r * KS + c);
  async_copy_b128(hid + (size_t)(m_blk + 64 + r) * FDIM + k0 + c, sA + (64 + r) * KS + c);
  async_copy_b128(w2b + (size_t)(n_blk + r) * FDIM + k0 + c,      sB + r * KS + c);
}

// ---------------------------------------------------------------------------
// FFN stage 1 per expert: hidden = silu(x @ w1^T) * (x @ v1^T)  (bf16 out).
// 256 thr = 8 wave32 in 4x2; each wave 32x32 (2x2 WMMA tiles) for g AND v.
// Double-buffered async global->LDS staging overlapped with WMMA.
// ---------------------------------------------------------------------------
__global__ __launch_bounds__(256) void moe_ffn1(const bf16_t* __restrict__ xb,
                                                const bf16_t* __restrict__ w1b,
                                                bf16_t* __restrict__ hidden) {
  __shared__ bf16_t sA[2][BM * KS];     // 2 x 8 KB
  __shared__ bf16_t sBg[2][BN * KS];    // 2 x 4 KB
  __shared__ bf16_t sBv[2][BN * KS];    // 2 x 4 KB

  const int tid  = threadIdx.x;
  const int lane = tid & 31;
  const int wave = tid >> 5;
  const int m_blk = blockIdx.x * BM;
  const int n_blk = blockIdx.y * BN;
  const int wm = (wave >> 1) * 32;      // wave row offset inside block tile
  const int wn = (wave & 1) * 32;       // wave col offset inside block tile

  const bf16_t* __restrict__ wg = w1b;                        // rows [0, F)
  const bf16_t* __restrict__ wv = w1b + (size_t)FDIM * HDIM;  // rows [F, 2F)

  v8f accg[2][2] = {};
  v8f accv[2][2] = {};

  stage_ffn1(xb, wg, wv, sA[0], sBg[0], sBv[0], m_blk, n_blk, 0, tid);

  int buf = 0;
  for (int k0 = 0; k0 < HDIM; k0 += KS) {
    wait_async();          // own async copies for sX[buf] have landed
    __syncthreads();       // ... and everyone else's too
    if (k0 + KS < HDIM)    // kick off next tile into the other buffer
      stage_ffn1(xb, wg, wv, sA[buf ^ 1], sBg[buf ^ 1], sBv[buf ^ 1],
                 m_blk, n_blk, k0 + KS, tid);

    v16bf a[2], bg[2], bv[2];
#pragma unroll
    for (int mi = 0; mi < 2; ++mi) a[mi]  = lds_a_frag(sA[buf],  wm + 16 * mi, lane);
#pragma unroll
    for (int ni = 0; ni < 2; ++ni) {
      bg[ni] = lds_b_frag(sBg[buf], wn + 16 * ni, lane);
      bv[ni] = lds_b_frag(sBv[buf], wn + 16 * ni, lane);
    }
#pragma unroll
    for (int mi = 0; mi < 2; ++mi)
#pragma unroll
      for (int ni = 0; ni < 2; ++ni) {
        accg[mi][ni] = wmma_bf16(a[mi], bg[ni], accg[mi][ni]);
        accv[mi][ni] = wmma_bf16(a[mi], bv[ni], accv[mi][ni]);
      }
    buf ^= 1;
    __syncthreads();       // compute done before next iter overwrites old buf
  }

  // Fused SwiGLU epilogue. C layout: VGPR r -> M = r + 8*(lane>=16), N = lane&15.
#pragma unroll
  for (int mi = 0; mi < 2; ++mi)
#pragma unroll
    for (int ni = 0; ni < 2; ++ni) {
      const int col = n_blk + wn + 16 * ni + (lane & 15);
#pragma unroll
      for (int r = 0; r < 8; ++r) {
        const int row = m_blk + wm + 16 * mi + ((lane >> 4) << 3) + r;
        const float g = accg[mi][ni][r];
        const float v = accv[mi][ni][r];
        const float sig = __builtin_amdgcn_rcpf(1.0f + __expf(-g));
        hidden[(size_t)row * FDIM + col] = f2bf(g * sig * v);
      }
    }
}

// ---------------------------------------------------------------------------
// FFN stage 2 per expert: out += combine[:, e] * (hidden @ w2^T)
// ---------------------------------------------------------------------------
__global__ __launch_bounds__(256) void moe_ffn2(const bf16_t* __restrict__ hidden,
                                                const bf16_t* __restrict__ w2b,
                                                const float* __restrict__ combine,
                                                int e,
                                                float* __restrict__ out) {
  __shared__ bf16_t sA[2][BM * KS];     // 2 x 8 KB
  __shared__ bf16_t sB[2][BN * KS];     // 2 x 4 KB

  const int tid  = threadIdx.x;
  const int lane = tid & 31;
  const int wave = tid >> 5;
  const int m_blk = blockIdx.x * BM;
  const int n_blk = blockIdx.y * BN;
  const int wm = (wave >> 1) * 32;
  const int wn = (wave & 1) * 32;

  v8f acc[2][2] = {};

  stage_ffn2(hidden, w2b, sA[0], sB[0], m_blk, n_blk, 0, tid);

  int buf = 0;
  for (int k0 = 0; k0 < FDIM; k0 += KS) {
    wait_async();
    __syncthreads();
    if (k0 + KS < FDIM)
      stage_ffn2(hidden, w2b, sA[buf ^ 1], sB[buf ^ 1], m_blk, n_blk, k0 + KS, tid);

    v16bf a[2], b[2];
#pragma unroll
    for (int mi = 0; mi < 2; ++mi) a[mi] = lds_a_frag(sA[buf], wm + 16 * mi, lane);
#pragma unroll
    for (int ni = 0; ni < 2; ++ni) b[ni] = lds_b_frag(sB[buf], wn + 16 * ni, lane);
#pragma unroll
    for (int mi = 0; mi < 2; ++mi)
#pragma unroll
      for (int ni = 0; ni < 2; ++ni)
        acc[mi][ni] = wmma_bf16(a[mi], b[ni], acc[mi][ni]);
    buf ^= 1;
    __syncthreads();
  }

#pragma unroll
  for (int mi = 0; mi < 2; ++mi)
#pragma unroll
    for (int ni = 0; ni < 2; ++ni) {
      const int col = n_blk + wn + 16 * ni + (lane & 15);
#pragma unroll
      for (int r = 0; r < 8; ++r) {
        const int row = m_blk + wm + 16 * mi + ((lane >> 4) << 3) + r;
        const float cw = combine[(size_t)row * NEXP + e];
        out[(size_t)row * HDIM + col] += cw * acc[mi][ni][r];
      }
    }
}

// ---------------------------------------------------------------------------
// Launch pipeline. Workspace layout:
//   combine 64KB | xb 4MB | hidden 4MB | w1b 4MB | w2b 2MB   (~14.1 MB)
// Weights converted to bf16 per expert (read-once stream), so GEMM loops
// touch only bf16 on both operands.
// ---------------------------------------------------------------------------
extern "C" void kernel_launch(void* const* d_in, const int* in_sizes, int n_in,
                              void* d_out, int out_size, void* d_ws, size_t ws_size,
                              hipStream_t stream) {
  const float* x      = (const float*)d_in[0];
  const float* gate_w = (const float*)d_in[1];
  const float* w1v1   = (const float*)d_in[2];
  const float* w2     = (const float*)d_in[3];
  float* out = (float*)d_out;

  char* ws = (char*)d_ws;
  size_t off = 0;
  float*  combine = (float*)(ws + off);  off += (size_t)T_TOK * NEXP * 4;  // 64 KB
  bf16_t* xb      = (bf16_t*)(ws + off); off += (size_t)T_TOK * HDIM * 2;  // 4 MB
  bf16_t* hidden  = (bf16_t*)(ws + off); off += (size_t)T_TOK * FDIM * 2;  // 4 MB
  bf16_t* w1b     = (bf16_t*)(ws + off); off += (size_t)2 * FDIM * HDIM * 2; // 4 MB
  bf16_t* w2b     = (bf16_t*)(ws + off);                                     // 2 MB

  moe_router<<<T_TOK / 256, 256, 0, stream>>>(x, gate_w, combine);
  f32_to_bf16_vec<<<(T_TOK * HDIM / 8) / 256, 256, 0, stream>>>(x, xb);
  zero_f32<<<(T_TOK * HDIM / 4) / 256, 256, 0, stream>>>(out);

  dim3 blk(256);
  dim3 g1(T_TOK / BM, FDIM / BN);
  dim3 g2(T_TOK / BM, HDIM / BN);
  for (int e = 0; e < NEXP; ++e) {
    f32_to_bf16_vec<<<(2 * FDIM * HDIM / 8) / 256, 256, 0, stream>>>(
        w1v1 + (size_t)e * 2 * FDIM * HDIM, w1b);
    f32_to_bf16_vec<<<(HDIM * FDIM / 8) / 256, 256, 0, stream>>>(
        w2 + (size_t)e * HDIM * FDIM, w2b);
    moe_ffn1<<<g1, blk, 0, stream>>>(xb, w1b, hidden);
    moe_ffn2<<<g2, blk, 0, stream>>>(hidden, w2b, combine, e, out);
  }
}
